// Classifier_65884798320805
// MI455X (gfx1250) — matrix operand order repaired and checked by reading, not compiled, not run on hardware
//
#include <hip/hip_runtime.h>
#include <hip/hip_bf16.h>
#include <math.h>

typedef __bf16 bf16_t;
typedef __attribute__((ext_vector_type(16))) __bf16 v16bf;
typedef __attribute__((ext_vector_type(8)))  __bf16 v8bf;
typedef __attribute__((ext_vector_type(8)))  float  v8f;

struct Frag { v8bf lo, hi; };

__device__ __forceinline__ v16bf frag_cat(const Frag& f) {
    v16bf v;
#pragma unroll
    for (int i = 0; i < 8; ++i) { v[i] = f.lo[i]; v[i + 8] = f.hi[i]; }
    return v;
}

// ---------------------------------------------------------------------------
// f32 -> bf16 conversion / fill
// ---------------------------------------------------------------------------
__global__ void cvt_bf16_kernel(const float* __restrict__ in, bf16_t* __restrict__ out, long n) {
    long tid = (long)blockIdx.x * blockDim.x + threadIdx.x;
    if (tid < n) out[tid] = (bf16_t)in[tid];
}

__global__ void fill_f32_kernel(float* __restrict__ p, float v, long n) {
    long tid = (long)blockIdx.x * blockDim.x + threadIdx.x;
    if (tid < n) p[tid] = v;
}

// ---------------------------------------------------------------------------
// bf16 WMMA GEMM:  C[M,Nout] = A[M,K] * W[Nout,K]^T   (f32 accumulate)
// One wave owns a 32(M) x 64(N) tile: 2 A-frags + 4 B-frags -> 8 WMMAs per
// 32-deep K-step. Fragments for step kb+32 are loaded before the WMMAs of
// step kb (manual double buffering) so loads overlap matrix math.
// Fragment layouts per CDNA5 ISA 7.12.2 (wave32).
// ---------------------------------------------------------------------------
__global__ __launch_bounds__(256)
void gemm_bf16_wmma(const bf16_t* __restrict__ A, const bf16_t* __restrict__ W,
                    float* __restrict__ C, int M, int Nout, int K, int ldc) {
    const int lane = threadIdx.x & 31;
    const int half = lane >> 4;     // 0: lanes 0-15, 1: lanes 16-31
    const int r16  = lane & 15;

    int wid     = (int)((blockIdx.x * blockDim.x + threadIdx.x) >> 5);
    int nChunks = Nout >> 6;
    int total   = (M >> 5) * nChunks;
    if (wid >= total) return;
    int mBase = (wid / nChunks) << 5;
    int nBase = (wid % nChunks) << 6;

    // A-frag 16x32: lane<16 -> K {kb..kb+7, kb+16..kb+23}; lanes 16-31 -> +8
    const bf16_t* arow0 = A + (size_t)(mBase + r16) * K + half * 8;
    const bf16_t* arow1 = A + (size_t)(mBase + 16 + r16) * K + half * 8;
    // B-frag 32x16 of W^T: lane n holds W row (nBase+16j+n),
// K {kb..kb+7, kb+16..kb+23} for lanes 0-15, +8 for lanes 16-31.
    const bf16_t* wrow = W + (size_t)(nBase + r16) * K + half * 8;
    const size_t  wstep = (size_t)16 * K;   // next 16 output columns

    auto loadA = [&](const bf16_t* p, int kb) {
        Frag f;
        f.lo = *(const v8bf*)(p + kb);
        f.hi = *(const v8bf*)(p + kb + 16);
        return f;
    };
    auto loadB = [&](int j, int kb) {
        Frag f;
        const bf16_t* p = wrow + (size_t)j * wstep + kb;
        f.lo = *(const v8bf*)(p);
        f.hi = *(const v8bf*)(p + 16);
        return f;
    };

    v8f acc[8] = {};   // acc[j]: rows 0-15 x cols 16j..16j+15 ; acc[4+j]: rows 16-31

    Frag a0c = loadA(arow0, 0);
    Frag a1c = loadA(arow1, 0);
    Frag bc[4];
#pragma unroll
    for (int j = 0; j < 4; ++j) bc[j] = loadB(j, 0);

    for (int kb = 0; kb < K - 32; kb += 32) {
        // prefetch next K-step into registers (independent of current WMMAs)
        Frag a0n = loadA(arow0, kb + 32);
        Frag a1n = loadA(arow1, kb + 32);
        Frag bn[4];
#pragma unroll
        for (int j = 0; j < 4; ++j) bn[j] = loadB(j, kb + 32);

        v16bf a0 = frag_cat(a0c), a1 = frag_cat(a1c);
#pragma unroll
        for (int j = 0; j < 4; ++j) {
            v16bf b = frag_cat(bc[j]);
            acc[j]     = __builtin_amdgcn_wmma_f32_16x16x32_bf16(
                false, a0, false, b, (short)0, acc[j],     false, false);
            acc[4 + j] = __builtin_amdgcn_wmma_f32_16x16x32_bf16(
                false, a1, false, b, (short)0, acc[4 + j], false, false);
        }
        a0c = a0n; a1c = a1n;
#pragma unroll
        for (int j = 0; j < 4; ++j) bc[j] = bn[j];
    }
    {   // last K-step
        v16bf a0 = frag_cat(a0c), a1 = frag_cat(a1c);
#pragma unroll
        for (int j = 0; j < 4; ++j) {
            v16bf b = frag_cat(bc[j]);
            acc[j]     = __builtin_amdgcn_wmma_f32_16x16x32_bf16(
                false, a0, false, b, (short)0, acc[j],     false, false);
            acc[4 + j] = __builtin_amdgcn_wmma_f32_16x16x32_bf16(
                false, a1, false, b, (short)0, acc[4 + j], false, false);
        }
    }

    // C/D layout: VGPR r -> M = r + 8*half ; N = lane&15
#pragma unroll
    for (int g = 0; g < 2; ++g) {
#pragma unroll
        for (int j = 0; j < 4; ++j) {
            float* cp = C + (size_t)ldc * (mBase + g * 16 + half * 8) + nBase + j * 16 + r16;
#pragma unroll
            for (int rr = 0; rr < 8; ++rr) cp[(size_t)ldc * rr] = acc[g * 4 + j][rr];
        }
    }
}

// ---------------------------------------------------------------------------
// el/er: per (row, head) wave-reduced dot of feat row-slice with al/ar
// ---------------------------------------------------------------------------
__global__ void dots_kernel(const float* __restrict__ G, const float* __restrict__ al,
                            const float* __restrict__ ar, float* __restrict__ el,
                            float* __restrict__ er, int rows, int ldg) {
    int wid  = (int)((blockIdx.x * blockDim.x + threadIdx.x) >> 5);
    int lane = threadIdx.x & 31;
    if (wid >= rows * 3) return;
    int row = wid / 3, h = wid % 3;
    const float* g = G + (size_t)row * ldg + h * 256 + lane * 8;
    const float* a = al + h * 256 + lane * 8;
    const float* b = ar + h * 256 + lane * 8;
    float sl = 0.f, sr = 0.f;
#pragma unroll
    for (int i = 0; i < 8; ++i) { float v = g[i]; sl += v * a[i]; sr += v * b[i]; }
    for (int off = 16; off; off >>= 1) {
        sl += __shfl_xor(sl, off, 32);
        sr += __shfl_xor(sr, off, 32);
    }
    if (!lane) { el[wid] = sl; er[wid] = sr; }
}

// ---------------------------------------------------------------------------
// Edge softmax pass 1: e = leaky_relu(el[src]+er[dst]); segment max on dst
// ---------------------------------------------------------------------------
__global__ void edge_max_kernel(const float* __restrict__ el, const float* __restrict__ er,
                                const int* __restrict__ src, const int* __restrict__ dst,
                                float* __restrict__ ebuf, float* __restrict__ mbuf,
                                int E, int C) {
    int tid = blockIdx.x * blockDim.x + threadIdx.x;
    if (tid >= E * C) return;
    int e = tid / C, c = tid - e * C;
    float v = el[(size_t)src[e] * C + c] + er[(size_t)dst[e] * C + c];
    v = v > 0.f ? v : 0.2f * v;
    ebuf[tid] = v;
    float* addr = mbuf + (size_t)dst[e] * C + c;
    if (v >= 0.f) atomicMax((int*)addr, __float_as_int(v));
    else          atomicMin((unsigned int*)addr, (unsigned int)__float_as_int(v));
}

// pass 2: p = exp(e - m[dst]); segment sum on dst
__global__ void edge_psum_kernel(float* __restrict__ ebuf, const float* __restrict__ mbuf,
                                 float* __restrict__ sbuf, const int* __restrict__ dst,
                                 int E, int C) {
    int tid = blockIdx.x * blockDim.x + threadIdx.x;
    if (tid >= E * C) return;
    int e = tid / C, c = tid - e * C;
    size_t di = (size_t)dst[e] * C + c;
    float p = expf(ebuf[tid] - mbuf[di]);
    ebuf[tid] = p;
    atomicAdd(sbuf + di, p);
}

// pass 3: acc[dst] += (p/s[dst]) * feat[src]  (4 floats per thread)
__global__ void edge_msg_kernel(const float* __restrict__ ebuf, const float* __restrict__ sbuf,
                                const int* __restrict__ src, const int* __restrict__ dst,
                                const float* __restrict__ feat, float* __restrict__ acc,
                                int E, int C, int P) {
    long tid = (long)blockIdx.x * blockDim.x + threadIdx.x;
    if (tid >= (long)E * C * 64) return;
    int dq = (int)(tid & 63);
    long t = tid >> 6;
    int c = (int)(t % C);
    int e = (int)(t / C);
    int s = src[e], d = dst[e];
    float alpha = ebuf[(size_t)e * C + c] / sbuf[(size_t)d * C + c];
    int pp = c / 3, h = c - pp * 3;
    const float* fp = feat + ((size_t)s * P + pp) * 1536 + h * 256 + dq * 4;
    float*       ap = acc  + ((size_t)d * P + pp) * 768  + h * 256 + dq * 4;
#pragma unroll
    for (int i = 0; i < 4; ++i) atomicAdd(ap + i, alpha * fp[i]);
}

// ---------------------------------------------------------------------------
// finalize: h = elu(acc + residual(G cols 768..1535) + bias) -> bf16
// ---------------------------------------------------------------------------
__global__ void finalize_kernel(const float* __restrict__ acc, const float* __restrict__ G,
                                const float* __restrict__ bias, bf16_t* __restrict__ out,
                                long total) {
    long tid = (long)blockIdx.x * blockDim.x + threadIdx.x;
    if (tid >= total) return;
    long row = tid / 768;
    int  j   = (int)(tid - row * 768);
    float v = acc[tid] + G[(size_t)row * 1536 + 768 + j] + bias[j];
    v = v > 0.f ? v : expm1f(v);
    out[tid] = (bf16_t)v;
}

// ---------------------------------------------------------------------------
// head: out[n,o] = sum_k relu(fc1out[n,k]+fc1b[k]) * fc2W[o,k] + fc2b[o]
// ---------------------------------------------------------------------------
__global__ void head_kernel(const float* __restrict__ fc1out, const float* __restrict__ fc1b,
                            const float* __restrict__ fc2W, const float* __restrict__ fc2b,
                            float* __restrict__ out, int N) {
    int tid = blockIdx.x * blockDim.x + threadIdx.x;
    if (tid >= N * 10) return;
    int n = tid / 10, o = tid - n * 10;
    const float* v = fc1out + (size_t)n * 64;
    const float* w = fc2W + o * 64;
    float s = fc2b[o];
#pragma unroll
    for (int k = 0; k < 64; ++k) {
        float h = v[k] + fc1b[k];
        s += (h > 0.f ? h : 0.f) * w[k];
    }
    out[tid] = s;
}

// ---------------------------------------------------------------------------
extern "C" void kernel_launch(void* const* d_in, const int* in_sizes, int n_in,
                              void* d_out, int out_size, void* d_ws, size_t ws_size,
                              hipStream_t stream) {
    (void)n_in; (void)out_size; (void)ws_size;
    const float* x    = (const float*)d_in[0];
    const int*   src  = (const int*)d_in[1];
    const int*   dst  = (const int*)d_in[2];
    const float* W1   = (const float*)d_in[3];
    const float* al1  = (const float*)d_in[4];
    const float* ar1  = (const float*)d_in[5];
    const float* rW1  = (const float*)d_in[6];
    const float* b1   = (const float*)d_in[7];
    const float* W2   = (const float*)d_in[8];
    const float* al2  = (const float*)d_in[9];
    const float* ar2  = (const float*)d_in[10];
    const float* rW2  = (const float*)d_in[11];
    const float* b2   = (const float*)d_in[12];
    const float* fc1W = (const float*)d_in[13];
    const float* fc1b = (const float*)d_in[14];
    const float* fc2W = (const float*)d_in[15];
    const float* fc2b = (const float*)d_in[16];

    const int N = in_sizes[0] / 256;   // nodes
    const int E = in_sizes[1];         // edges

    // ---- workspace layout (bump allocator, 256B aligned) ----
    char*  base = (char*)d_ws;
    size_t cur  = 0;
    auto take = [&](size_t bytes) -> char* {
        cur = (cur + 255) & ~(size_t)255;
        char* p = base + cur;
        cur += bytes;
        return p;
    };
    bf16_t* wcat1 = (bf16_t*)take((size_t)1536 * 256 * 2);
    bf16_t* wcat2 = (bf16_t*)take((size_t)1536 * 256 * 2);
    bf16_t* fcWb  = (bf16_t*)take((size_t)64 * 2304 * 2);
    float*  elb   = (float*)take((size_t)N * 9 * 4);
    float*  erb   = (float*)take((size_t)N * 9 * 4);
    float*  mb    = (float*)take((size_t)N * 9 * 4);
    float*  sb    = (float*)take((size_t)N * 9 * 4);
    float*  ebuf  = (float*)take((size_t)E * 9 * 4);
    bf16_t* h1b   = (bf16_t*)take((size_t)N * 3 * 256 * 2);
    bf16_t* h2b   = (bf16_t*)take((size_t)N * 2304 * 2);
    float*  fc1o  = (float*)take((size_t)N * 64 * 4);
    float*  accb  = (float*)take((size_t)N * 2304 * 4);
    // G2 reuses the (xb, G1) region: both are dead before G2 is written.
    size_t xb_b  = ((size_t)N * 256 * 2 + 255) & ~(size_t)255;
    size_t G1_b  = (size_t)N * 1536 * 4;
    size_t G2_b  = (size_t)N * 3 * 1536 * 4;
    size_t big_b = (xb_b + G1_b > G2_b) ? (xb_b + G1_b) : G2_b;
    char*  big   = take(big_b);
    bf16_t* xb = (bf16_t*)big;
    float*  G1 = (float*)(big + xb_b);
    float*  G2 = (float*)big;

    auto cdiv = [](long a, long b) { return (unsigned int)((a + b - 1) / b); };
    const int BT = 256;

    // ---- convert inputs / weights to bf16 ----
    cvt_bf16_kernel<<<cdiv((long)N * 256, BT), BT, 0, stream>>>(x, xb, (long)N * 256);
    cvt_bf16_kernel<<<cdiv(768 * 256, BT), BT, 0, stream>>>(W1,  wcat1,            768 * 256);
    cvt_bf16_kernel<<<cdiv(768 * 256, BT), BT, 0, stream>>>(rW1, wcat1 + 768 * 256, 768 * 256);
    cvt_bf16_kernel<<<cdiv(768 * 256, BT), BT, 0, stream>>>(W2,  wcat2,            768 * 256);
    cvt_bf16_kernel<<<cdiv(768 * 256, BT), BT, 0, stream>>>(rW2, wcat2 + 768 * 256, 768 * 256);
    cvt_bf16_kernel<<<cdiv(64 * 2304, BT), BT, 0, stream>>>(fc1W, fcWb, 64 * 2304);

    // ---- layer 1 ----
    {
        long waves = (long)(N / 32) * (1536 / 64);
        gemm_bf16_wmma<<<cdiv(waves * 32, BT), BT, 0, stream>>>(xb, wcat1, G1, N, 1536, 256, 1536);
    }
    dots_kernel<<<cdiv((long)N * 3 * 32, BT), BT, 0, stream>>>(G1, al1, ar1, elb, erb, N, 1536);
    fill_f32_kernel<<<cdiv((long)N * 3, BT), BT, 0, stream>>>(mb, -__builtin_inff(), (long)N * 3);
    fill_f32_kernel<<<cdiv((long)N * 3, BT), BT, 0, stream>>>(sb, 0.f, (long)N * 3);
    fill_f32_kernel<<<cdiv((long)N * 768, BT), BT, 0, stream>>>(accb, 0.f, (long)N * 768);
    edge_max_kernel<<<cdiv((long)E * 3, BT), BT, 0, stream>>>(elb, erb, src, dst, ebuf, mb, E, 3);
    edge_psum_kernel<<<cdiv((long)E * 3, BT), BT, 0, stream>>>(ebuf, mb, sb, dst, E, 3);
    edge_msg_kernel<<<cdiv((long)E * 3 * 64, BT), BT, 0, stream>>>(ebuf, sb, src, dst, G1, accb, E, 3, 1);
    finalize_kernel<<<cdiv((long)N * 768, BT), BT, 0, stream>>>(accb, G1, b1, h1b, (long)N * 768);

    // ---- layer 2 (M = 3N rows) ----
    {
        long waves = (long)(N * 3 / 32) * (1536 / 64);
        gemm_bf16_wmma<<<cdiv(waves * 32, BT), BT, 0, stream>>>(h1b, wcat2, G2, N * 3, 1536, 256, 1536);
    }
    dots_kernel<<<cdiv((long)N * 9 * 32, BT), BT, 0, stream>>>(G2, al2, ar2, elb, erb, N * 3, 1536);
    fill_f32_kernel<<<cdiv((long)N * 9, BT), BT, 0, stream>>>(mb, -__builtin_inff(), (long)N * 9);
    fill_f32_kernel<<<cdiv((long)N * 9, BT), BT, 0, stream>>>(sb, 0.f, (long)N * 9);
    fill_f32_kernel<<<cdiv((long)N * 2304, BT), BT, 0, stream>>>(accb, 0.f, (long)N * 2304);
    edge_max_kernel<<<cdiv((long)E * 9, BT), BT, 0, stream>>>(elb, erb, src, dst, ebuf, mb, E, 9);
    edge_psum_kernel<<<cdiv((long)E * 9, BT), BT, 0, stream>>>(ebuf, mb, sb, dst, E, 9);
    edge_msg_kernel<<<cdiv((long)E * 9 * 64, BT), BT, 0, stream>>>(ebuf, sb, src, dst, G2, accb, E, 9, 3);
    finalize_kernel<<<cdiv((long)N * 2304, BT), BT, 0, stream>>>(accb, G2, b2, h2b, (long)N * 2304);

    // ---- MLP head ----
    {
        long waves = (long)(N / 32) * (64 / 64);
        gemm_bf16_wmma<<<cdiv(waves * 32, BT), BT, 0, stream>>>(h2b, fcWb, fc1o, N, 64, 2304, 64);
    }
    head_kernel<<<cdiv((long)N * 10, BT), BT, 0, stream>>>(fc1o, fc1b, fc2W, fc2b, (float*)d_out, N);
}
